// BaseGenerator_55963423867464
// MI455X (gfx1250) — matrix operand order repaired and checked by reading, not compiled, not run on hardware
//
#include <hip/hip_runtime.h>
#include <hip/hip_bf16.h>

#define BB 4
#define LL 1024
#define EE 1024
#define HH 16
#define DH 64
#define FFD 4096
#define NLAYERS 6
#define VV 140
#define BL (BB * LL)   // 4096

typedef __attribute__((ext_vector_type(16))) __bf16 v16bf;
typedef __attribute__((ext_vector_type(2)))  __bf16 v2bf;
typedef __attribute__((ext_vector_type(8)))  float  v8f;
typedef __attribute__((ext_vector_type(4)))  unsigned v4u;
typedef __attribute__((ext_vector_type(8)))  int    v8i;
typedef __attribute__((ext_vector_type(4)))  int    v4i;

union Frag {
    v16bf v;
    unsigned u[8];
};

// ---- fp32 -> bf16 helpers (hardware v_cvt_pk_bf16_f32 when available) ----
__device__ inline unsigned pack2(float lo, float hi) {
#if __has_builtin(__builtin_amdgcn_cvt_pk_bf16_f32)
    union { v2bf v; unsigned u; } r;
    r.v = __builtin_amdgcn_cvt_pk_bf16_f32(lo, hi);
    return r.u;
#else
    unsigned a = __float_as_uint(lo), b = __float_as_uint(hi);
    a = (a + 0x7FFFu + ((a >> 16) & 1u)) >> 16;
    b = (b + 0x7FFFu + ((b >> 16) & 1u)) >> 16;
    return a | (b << 16);
#endif
}
__device__ inline unsigned short f2bf(float f) {
    return (unsigned short)(pack2(f, 0.0f) & 0xFFFFu);
}

// A-matrix (16x32 bf16) pair base K for pair p (0..7), ISA 7.12.2:
// lanes 0-15: K = {0..7, 16..23}; lanes 16-31: K = {8..15, 24..31}
__device__ inline int a_kpair(int p, int lane) {
    int kh = (lane >> 4) * 8;
    return (p < 4 ? 2 * p : 16 + 2 * (p - 4)) + kh;
}

__device__ inline v8f wmma_bf16(const Frag& a, const Frag& b, v8f c) {
    return __builtin_amdgcn_wmma_f32_16x16x32_bf16(
        false, a.v, false, b.v, (short)0, c, false, false);
}

// ---- Tensor Data Mover: 2-D tile load, bf16 elements (data_size=2B) ----
// D# per CDNA5 ISA ch.8: group0 = count/lds/global/type, group1 = dims/strides.
__device__ inline void tdm_load_2d(const void* gptr, unsigned lds_off,
                                   unsigned tensor_d0, unsigned tensor_d1,
                                   unsigned tile_d0, unsigned tile_d1,
                                   unsigned stride_d0_elems) {
    unsigned long long ga = (unsigned long long)gptr;
    v4u g0;
    g0[0] = 1u;                                                // count=1 (valid), user mode
    g0[1] = lds_off;                                           // lds_addr (bytes)
    g0[2] = (unsigned)(ga & 0xFFFFFFFFu);                      // global_addr[31:0]
    g0[3] = (unsigned)((ga >> 32) & 0x01FFFFFFu) | (2u << 30); // addr[56:32] | type=2
    v8i g1;
    g1[0] = (int)(1u << 16);                                   // data_size=1 -> 2 bytes
    g1[1] = (int)((tensor_d0 & 0xFFFFu) << 16);                // tensor_dim0[15:0]
    g1[2] = (int)(((tensor_d0 >> 16) & 0xFFFFu) | ((tensor_d1 & 0xFFFFu) << 16));
    g1[3] = (int)(((tensor_d1 >> 16) & 0xFFFFu) | ((tile_d0 & 0xFFFFu) << 16));
    g1[4] = (int)(tile_d1 & 0xFFFFu);                          // tile_dim1 (tile_dim2=0)
    g1[5] = (int)stride_d0_elems;                              // tensor_dim0_stride[31:0]
    g1[6] = 0;
    g1[7] = 0;
    v4i z4 = {0, 0, 0, 0};
#if defined(__clang_major__) && __clang_major__ >= 23
    v8i z8 = {0, 0, 0, 0, 0, 0, 0, 0};
    __builtin_amdgcn_tensor_load_to_lds(g0, g1, z4, z4, z8, 0);
#else
    __builtin_amdgcn_tensor_load_to_lds(g0, g1, z4, z4, 0);
#endif
}

// ---------------------------------------------------------------- fp32 -> bf16 copy
__global__ __launch_bounds__(256) void k_f2bf(const float* __restrict__ in,
                                              unsigned* __restrict__ outp, long n2) {
    long i = (long)blockIdx.x * 256 + threadIdx.x;   // processes 2 elems
    if (i >= n2) return;
    const float2 f = ((const float2*)in)[i];
    outp[i] = pack2(f.x, f.y);
}

// ---------------------------------------------------------------- embedding (f32 + bf16)
__global__ __launch_bounds__(256) void k_embed(const int* __restrict__ seq,
                                               const float* __restrict__ temb,
                                               float* __restrict__ x,
                                               unsigned short* __restrict__ xb) {
    size_t i = (size_t)blockIdx.x * 256 + threadIdx.x;
    if (i >= (size_t)BL * EE) return;
    int bl = (int)(i >> 10);
    int e  = (int)(i & 1023);
    float v = temb[(size_t)seq[bl] * EE + e] * 32.0f;   // sqrt(1024)
    x[i]  = v;
    xb[i] = f2bf(v);
}

// ---------------------------------------------------------------- GEMM: C = A @ W^T + bias
// A: (M,K) bf16 row-major.  W: (N,K) bf16 row-major.  C: (M,N), ld = N.
// Block = 256 threads = 8 waves; block tile 32(M) x 128(N); wave = two 16x16 tiles.
// Double-buffered TDM pipeline: DMA of slab s+1 overlaps WMMA of slab s
// (TENSORcnt in-order completion; s_wait_tensorcnt(2) keeps the newest pair in flight).
// ACT: 0 none, 1 exact GELU.  OUTBF: 0 -> f32 C, 1 -> bf16 C.
template <int ACT, int OUTBF>
__global__ __launch_bounds__(256) void k_gemm_bf(const unsigned short* __restrict__ A,
                                                 const unsigned short* __restrict__ W,
                                                 const float* __restrict__ bias,
                                                 void* __restrict__ Cout,
                                                 int M, int N, int K) {
    __shared__ unsigned short sA[2][32 * 32];    // 2 x 2 KB
    __shared__ unsigned short sB[2][128 * 32];   // 2 x 8 KB
    const int t    = threadIdx.x;
    const int lane = t & 31;
    const int wave = t >> 5;
    const int m0   = blockIdx.y * 32;
    const int n0   = blockIdx.x * 128;

    const unsigned ldsA[2] = { (unsigned)(size_t)&sA[0][0], (unsigned)(size_t)&sA[1][0] };
    const unsigned ldsB[2] = { (unsigned)(size_t)&sB[0][0], (unsigned)(size_t)&sB[1][0] };
    const unsigned nRem = (unsigned)(N - n0);    // TDM OOB zero-fill covers N tail

    auto issue = [&](int step, int buf) {
        int k0 = step * 32;
        tdm_load_2d(A + (size_t)m0 * K + k0, ldsA[buf], 32u, 32u, 32u, 32u, (unsigned)K);
        tdm_load_2d(W + (size_t)n0 * K + k0, ldsB[buf], 32u, nRem, 32u, 128u, (unsigned)K);
    };

    v8f c0 = {}, c1 = {};
    const int steps = K / 32;

    if (wave == 0) issue(0, 0);

    for (int s = 0; s < steps; ++s) {
        const int buf = s & 1;
        if (wave == 0) {
            if (s + 1 < steps) {
                issue(s + 1, buf ^ 1);                    // prefetch next slab
                __builtin_amdgcn_s_wait_tensorcnt(2);     // current slab complete
            } else {
                __builtin_amdgcn_s_wait_tensorcnt(0);
            }
        }
        __syncthreads();                                  // slab visible to all waves

        Frag a0, a1, b;
        {
            int m = lane & 15;
#pragma unroll
            for (int p = 0; p < 8; ++p) {
                int kb = a_kpair(p, lane);
                a0.u[p] = *(const unsigned*)&sA[buf][m * 32 + kb];
                a1.u[p] = *(const unsigned*)&sA[buf][(16 + m) * 32 + kb];
            }
        }
        {
            int n = wave * 16 + (lane & 15);
#pragma unroll
            for (int p = 0; p < 8; ++p)
                b.u[p] = *(const unsigned*)&sB[buf][n * 32 + a_kpair(p, lane)];
        }
        c0 = wmma_bf16(a0, b, c0);
        c1 = wmma_bf16(a1, b, c1);
        __syncthreads();                                  // slab free for reuse
    }

    // epilogue: C/D layout — VGPR i: row = i + 8*(lane>=16), col = lane%16
    int n = n0 + wave * 16 + (lane & 15);
    if (n < N) {
        float bv = bias ? bias[n] : 0.0f;
#pragma unroll
        for (int i = 0; i < 8; ++i) {
            int   m = m0 + i + ((lane >> 4) << 3);
            float v0 = c0[i] + bv;
            float v1 = c1[i] + bv;
            if (ACT == 1) {
                v0 = 0.5f * v0 * (1.0f + erff(v0 * 0.70710678f));
                v1 = 0.5f * v1 * (1.0f + erff(v1 * 0.70710678f));
            }
            if (OUTBF) {
                ((unsigned short*)Cout)[(size_t)m * N + n]        = f2bf(v0);
                ((unsigned short*)Cout)[(size_t)(m + 16) * N + n] = f2bf(v1);
            } else {
                ((float*)Cout)[(size_t)m * N + n]        = v0;
                ((float*)Cout)[(size_t)(m + 16) * N + n] = v1;
            }
        }
    }
}

// ---------------------------------------------------------------- fused attention
// One wave per (b, h, 16-row q tile). qkv is bf16 (B,L,3E). Output bf16 (B,L,E).
__global__ __launch_bounds__(32) void k_attn(const unsigned short* __restrict__ qkv,
                                             const int* __restrict__ dsq,
                                             const int* __restrict__ upq,
                                             const int* __restrict__ dnq,
                                             const int* __restrict__ rtq,
                                             const float* __restrict__ demb,
                                             const float* __restrict__ uemb,
                                             const float* __restrict__ dnemb,
                                             const float* __restrict__ remb,
                                             const int* __restrict__ seq,
                                             unsigned short* __restrict__ out) {
    __shared__ float          sc[16][LL];   // 64 KB score strip
    __shared__ unsigned short sp[16][LL];   // 32 KB bf16 probability strip
    const int lane = threadIdx.x;
    const int qt   = blockIdx.x & 63;
    const int h    = (blockIdx.x >> 6) & 15;
    const int b    = blockIdx.x >> 10;
    const size_t rs = 3 * EE;               // qkv row stride (elements)
    const unsigned short* base = qkv + (size_t)b * LL * rs;

    // Q fragments (two 32-wide d-halves) directly from bf16 global: pairs contiguous
    Frag qa[2];
    {
        int m = lane & 15;
        const unsigned short* qr = base + (size_t)(qt * 16 + m) * rs + h * DH;
#pragma unroll
        for (int s = 0; s < 2; ++s)
#pragma unroll
            for (int p = 0; p < 8; ++p)
                qa[s].u[p] = *(const unsigned*)(qr + s * 32 + a_kpair(p, lane));
    }

    // S = Q K^T / sqrt(DH) + rel-pos bias, causal+pad masked, into LDS
    for (int j = 0; j < 64; ++j) {
        Frag kb[2];
        {
            int key = j * 16 + (lane & 15);
            const unsigned short* kr = base + (size_t)key * rs + EE + h * DH;
#pragma unroll
            for (int s = 0; s < 2; ++s)
#pragma unroll
                for (int p = 0; p < 8; ++p)
                    kb[s].u[p] = *(const unsigned*)(kr + s * 32 + a_kpair(p, lane));
        }
        v8f c = {};
        c = wmma_bf16(qa[0], kb[0], c);
        c = wmma_bf16(qa[1], kb[1], c);

        int  keyc = j * 16 + (lane & 15);
        bool pad  = (seq[b * LL + keyc] == 0);
#pragma unroll
        for (int i = 0; i < 8; ++i) {
            int qm = i + ((lane >> 4) << 3);
            int q  = qt * 16 + qm;
            float sv;
            if (keyc > q || pad) {
                sv = -__builtin_inff();
            } else {
                size_t qk = ((size_t)(b * LL + q)) * LL + keyc;
                float bias = demb[dsq[qk] * HH + h] + uemb[upq[qk] * HH + h]
                           + dnemb[dnq[qk] * HH + h] + remb[rtq[qk] * HH + h];
                sv = c[i] * 0.125f + bias;   // 1/sqrt(64)
            }
            sc[qm][keyc] = sv;
        }
    }
    __syncthreads();

    // row softmax (wave32 shuffle reductions); probabilities stored as bf16
    for (int r = 0; r < 16; ++r) {
        float mx = -__builtin_inff();
        for (int cc = lane; cc < LL; cc += 32) mx = fmaxf(mx, sc[r][cc]);
#pragma unroll
        for (int o = 16; o > 0; o >>= 1) mx = fmaxf(mx, __shfl_xor(mx, o, 32));
        float sum = 0.0f;
        for (int cc = lane; cc < LL; cc += 32) {
            float e = __expf(sc[r][cc] - mx);
            sc[r][cc] = e;
            sum += e;
        }
#pragma unroll
        for (int o = 16; o > 0; o >>= 1) sum += __shfl_xor(sum, o, 32);
        float inv = 1.0f / sum;
        for (int cc = lane; cc < LL; cc += 32)
            sp[r][cc] = f2bf(sc[r][cc] * inv);
    }
    __syncthreads();

    // O = P @ V : A fragments straight from bf16 LDS, 32-key steps, 4 d-tiles
    v8f acc[4] = {{}, {}, {}, {}};
    for (int ks = 0; ks < 32; ++ks) {
        Frag pa;
        {
            int mm = lane & 15;
#pragma unroll
            for (int p = 0; p < 8; ++p)
                pa.u[p] = *(const unsigned*)&sp[mm][ks * 32 + a_kpair(p, lane)];
        }
#pragma unroll
        for (int dt = 0; dt < 4; ++dt) {
            Frag vb;
            int d = dt * 16 + (lane & 15);
#pragma unroll
            for (int p = 0; p < 8; ++p) {
                int key = ks * 32 + a_kpair(p, lane);
                const unsigned short* v0 = base + (size_t)key * rs + 2 * EE + h * DH + d;
                vb.u[p] = (unsigned)v0[0] | ((unsigned)v0[rs] << 16);
            }
            acc[dt] = wmma_bf16(pa, vb, acc[dt]);
        }
    }
#pragma unroll
    for (int dt = 0; dt < 4; ++dt)
#pragma unroll
        for (int i = 0; i < 8; ++i) {
            int qm = i + ((lane >> 4) << 3);
            int q  = qt * 16 + qm;
            int d  = dt * 16 + (lane & 15);
            out[((size_t)(b * LL + q)) * EE + h * DH + d] = f2bf(acc[dt][i]);
        }
}

// ---------------------------------------------------------------- residual + LayerNorm
// out (f32) and outb (bf16 copy feeding the next GEMM's A operand)
__global__ __launch_bounds__(256) void k_add_ln(const float* __restrict__ hin,
                                                const float* __restrict__ delta,
                                                const float* __restrict__ g,
                                                const float* __restrict__ bt,
                                                float* __restrict__ out,
                                                unsigned short* __restrict__ outb) {
    __shared__ float red[256];
    const int row = blockIdx.x;
    const int t   = threadIdx.x;
    const float* hr = hin + (size_t)row * EE;
    const float* dr = delta ? delta + (size_t)row * EE : nullptr;

    float y[4];
    float s = 0.0f;
#pragma unroll
    for (int i = 0; i < 4; ++i) {
        int e = t + i * 256;
        y[i]  = hr[e] + (dr ? dr[e] : 0.0f);
        s += y[i];
    }
    red[t] = s;
    __syncthreads();
    for (int o = 128; o > 0; o >>= 1) {
        if (t < o) red[t] += red[t + o];
        __syncthreads();
    }
    float mean = red[0] * (1.0f / EE);
    __syncthreads();

    float v = 0.0f;
#pragma unroll
    for (int i = 0; i < 4; ++i) {
        float d0 = y[i] - mean;
        v += d0 * d0;
    }
    red[t] = v;
    __syncthreads();
    for (int o = 128; o > 0; o >>= 1) {
        if (t < o) red[t] += red[t + o];
        __syncthreads();
    }
    float inv = rsqrtf(red[0] * (1.0f / EE) + 1e-5f);
#pragma unroll
    for (int i = 0; i < 4; ++i) {
        int e = t + i * 256;
        float r = (y[i] - mean) * inv * g[e] + bt[e];
        out[(size_t)row * EE + e]  = r;
        outb[(size_t)row * EE + e] = f2bf(r);
    }
}

// ---------------------------------------------------------------- pred_mask -> -inf
__global__ __launch_bounds__(256) void k_mask(const unsigned char* __restrict__ pm,
                                              float* __restrict__ out, int n) {
    int i = blockIdx.x * 256 + threadIdx.x;
    if (i < n && pm[i]) out[i] = -__builtin_inff();
}

// ---------------------------------------------------------------- host driver
extern "C" void kernel_launch(void* const* d_in, const int* in_sizes, int n_in,
                              void* d_out, int out_size, void* d_ws, size_t ws_size,
                              hipStream_t stream) {
    const int*   seq   = (const int*)d_in[0];
    const int*   dsq   = (const int*)d_in[1];
    const int*   upq   = (const int*)d_in[2];
    const int*   dnq   = (const int*)d_in[3];
    const int*   rtq   = (const int*)d_in[4];
    const unsigned char* pm = (const unsigned char*)d_in[5];
    const float* temb  = (const float*)d_in[6];
    const float* demb  = (const float*)d_in[7];
    const float* uemb  = (const float*)d_in[8];
    const float* dnemb = (const float*)d_in[9];
    const float* remb  = (const float*)d_in[10];
    const float* Wqkv  = (const float*)d_in[11];
    const float* bqkv  = (const float*)d_in[12];
    const float* Wo    = (const float*)d_in[13];
    const float* bo    = (const float*)d_in[14];
    const float* W1    = (const float*)d_in[15];
    const float* b1    = (const float*)d_in[16];
    const float* W2    = (const float*)d_in[17];
    const float* b2    = (const float*)d_in[18];
    const float* ln1g  = (const float*)d_in[19];
    const float* ln1b  = (const float*)d_in[20];
    const float* ln2g  = (const float*)d_in[21];
    const float* ln2b  = (const float*)d_in[22];
    const float* flng  = (const float*)d_in[23];
    const float* flnb  = (const float*)d_in[24];
    const float* genW  = (const float*)d_in[25];
    const float* genb  = (const float*)d_in[26];
    float* out = (float*)d_out;

    // workspace layout
    char* w = (char*)d_ws;
    float*          h    = (float*)w;          w += (size_t)BL * EE * 4;          // f32 act
    float*          dl   = (float*)w;          w += (size_t)BL * EE * 4;          // f32 delta
    unsigned short* hb   = (unsigned short*)w; w += (size_t)BL * EE * 2;          // bf16 act
    unsigned short* qkvb = (unsigned short*)w; w += (size_t)BL * 3 * EE * 2;      // bf16 qkv
    unsigned short* aob  = (unsigned short*)w; w += (size_t)BL * EE * 2;          // bf16 attn out
    unsigned short* f1b  = (unsigned short*)w; w += (size_t)BL * FFD * 2;         // bf16 gelu out
    unsigned short* wb   = (unsigned short*)w;                                    // bf16 weights (max FFD*EE)

    k_embed<<<(BL * EE + 255) / 256, 256, 0, stream>>>(seq, temb, h, hb);

    for (int l = 0; l < NLAYERS; ++l) {
        const float* Wq  = Wqkv + (size_t)l * 3 * EE * EE;
        const float* bq  = bqkv + (size_t)l * 3 * EE;
        const float* Wol = Wo + (size_t)l * EE * EE;
        const float* bol = bo + (size_t)l * EE;
        const float* W1l = W1 + (size_t)l * FFD * EE;
        const float* b1l = b1 + (size_t)l * FFD;
        const float* W2l = W2 + (size_t)l * EE * FFD;
        const float* b2l = b2 + (size_t)l * EE;

        // QKV
        k_f2bf<<<(3 * EE * EE / 2 + 255) / 256, 256, 0, stream>>>(Wq, (unsigned*)wb, 3 * EE * EE / 2);
        k_gemm_bf<0, 1><<<dim3(3 * EE / 128, BL / 32), 256, 0, stream>>>(
            hb, wb, bq, qkvb, BL, 3 * EE, EE);
        // attention
        k_attn<<<BB * HH * (LL / 16), 32, 0, stream>>>(
            qkvb, dsq, upq, dnq, rtq, demb, uemb, dnemb, remb, seq, aob);
        // Wo projection
        k_f2bf<<<(EE * EE / 2 + 255) / 256, 256, 0, stream>>>(Wol, (unsigned*)wb, EE * EE / 2);
        k_gemm_bf<0, 0><<<dim3(EE / 128, BL / 32), 256, 0, stream>>>(
            aob, wb, bol, dl, BL, EE, EE);
        k_add_ln<<<BL, 256, 0, stream>>>(h, dl, ln1g + (size_t)l * EE,
                                         ln1b + (size_t)l * EE, h, hb);
        // FFN
        k_f2bf<<<(FFD * EE / 2 + 255) / 256, 256, 0, stream>>>(W1l, (unsigned*)wb, FFD * EE / 2);
        k_gemm_bf<1, 1><<<dim3(FFD / 128, BL / 32), 256, 0, stream>>>(
            hb, wb, b1l, f1b, BL, FFD, EE);
        k_f2bf<<<(EE * FFD / 2 + 255) / 256, 256, 0, stream>>>(W2l, (unsigned*)wb, EE * FFD / 2);
        k_gemm_bf<0, 0><<<dim3(EE / 128, BL / 32), 256, 0, stream>>>(
            f1b, wb, b2l, dl, BL, EE, FFD);
        k_add_ln<<<BL, 256, 0, stream>>>(h, dl, ln2g + (size_t)l * EE,
                                         ln2b + (size_t)l * EE, h, hb);
    }

    // final LN + vocab projection + mask
    k_add_ln<<<BL, 256, 0, stream>>>(h, nullptr, flng, flnb, h, hb);
    k_f2bf<<<(VV * EE / 2 + 255) / 256, 256, 0, stream>>>(genW, (unsigned*)wb, VV * EE / 2);
    k_gemm_bf<0, 0><<<dim3((VV + 127) / 128, BL / 32), 256, 0, stream>>>(
        hb, wb, genb, out, BL, VV, EE);
    k_mask<<<(BL * VV + 255) / 256, 256, 0, stream>>>(pm, out, BL * VV);
}
